// DecoderBlock_63720134803566
// MI455X (gfx1250) — matrix operand order repaired and checked by reading, not compiled, not run on hardware
//
#include <hip/hip_runtime.h>

// ---------------------------------------------------------------------------
// Decoder block (masked self-attn + cross-attn + FFN + 3 LayerNorms), gfx1250.
// GEMMs: v_wmma_f32_16x16x32_bf16, double-buffered LDS fed by
// global_load_async_to_lds_b128 (ASYNCcnt), fp32 accumulate.
// ---------------------------------------------------------------------------

typedef __bf16 bf16;
typedef __bf16 v16bf __attribute__((ext_vector_type(16)));
typedef float  v8f   __attribute__((ext_vector_type(8)));
typedef unsigned int u32;

#define SQ     4096
#define DMODEL 1024
#define NHEAD  16
#define DKH    64
#define DFF    4096

// ---------------------------------------------------------------- conversion
__global__ __launch_bounds__(256)
void cvt_f32_bf16(const float* __restrict__ in, bf16* __restrict__ out, int n) {
  int i = blockIdx.x * blockDim.x + threadIdx.x;
  int stride = gridDim.x * blockDim.x;
  for (; i < n; i += stride) out[i] = (bf16)in[i];
}

// ---------------------------------------------------------- 32x32 transpose
// in [R][C] -> outT [C][R]   (R, C multiples of 32)
__global__ __launch_bounds__(256)
void transpose_bf16(const bf16* __restrict__ in, bf16* __restrict__ outT,
                    int R, int C)
{
  __shared__ bf16 tile[32][33];
  int bx = blockIdx.x * 32;          // column origin in `in`
  int by = blockIdx.y * 32;          // row origin in `in`
  int tx = threadIdx.x & 31;
  int ty = threadIdx.x >> 5;         // 8 row-phases
  #pragma unroll
  for (int i = ty; i < 32; i += 8)
    tile[i][tx] = in[(size_t)(by + i) * C + bx + tx];
  __syncthreads();
  #pragma unroll
  for (int i = ty; i < 32; i += 8)
    outT[(size_t)(bx + i) * R + by + tx] = tile[tx][i];
}

// ------------------------------------------------------------------ WMMA GEMM
// C[M,N] = A[M,K] * B[N,K]^T (+bias)(+relu), bf16 in, fp32 accumulate.
// Block tile 128x128, K-step 32, double-buffered LDS stages filled with
// global_load_async_to_lds_b128. 8 wave32s arranged 4(m) x 2(n); each wave
// owns 32x64 = 2x4 v_wmma_f32_16x16x32_bf16 tiles.
// causal!=0: skip blocks entirely above the diagonal (consumer never reads).
__global__ __launch_bounds__(256)
void gemm_bf16_wmma(const bf16* __restrict__ A, int lda,
                    const bf16* __restrict__ B, int ldb,
                    const float* __restrict__ bias, int relu,
                    float* __restrict__ outF, bf16* __restrict__ outH, int ldc,
                    int M, int N, int K, int causal)
{
  const int m0 = blockIdx.y * 128;
  const int n0 = blockIdx.x * 128;
  if (causal && n0 >= m0 + 128) return;   // block-uniform early out

  // 2 stages x (A 128x32 bf16 = 8KB, B 128x32 bf16 = 8KB) = 32KB
  __shared__ __align__(16) u32 lds_u[8192];
  const u32 ldsBase = (u32)(uintptr_t)(void*)lds_u;  // LDS byte offset (addr[31:0])

  const int tid  = threadIdx.x;
  const int lane = tid & 31;
  const int wave = tid >> 5;
  const int wm   = wave >> 1;        // 0..3 -> m origin wm*32
  const int wn   = wave & 1;         // 0..1 -> n origin wn*64
  const int l16  = lane & 15;
  const int lhi  = lane >> 4;        // half-wave select (ISA §7.12.2)

  // Per-stage async fill: 4 instructions/wave (2 for A, 2 for B), 16B/lane.
  // B rows are clamped to N-1 so every lane always issues (uniform ASYNCcnt);
  // garbage columns only feed C columns >= N which are never stored.
  auto issue_stage = [&](int p, int k0) {
    const u32 aByte = ldsBase + (u32)p * 16384u;
    const u32 bByte = aByte + 8192u;
    #pragma unroll
    for (int j = 0; j < 2; j++) {
      int e   = tid + j * 256;       // 0..511
      int row = e >> 2;              // 0..127
      int cg  = e & 3;               // 16B chunk within 64B row
      const bf16* ga = A + (size_t)(m0 + row) * lda + k0 + cg * 8;
      u32 la = aByte + (u32)(row * 64 + cg * 16);
      asm volatile("global_load_async_to_lds_b128 %0, %1, off"
                   :: "v"(la), "v"(ga) : "memory");
      int rowc = n0 + row; if (rowc > N - 1) rowc = N - 1;
      const bf16* gb = B + (size_t)rowc * ldb + k0 + cg * 8;
      u32 lb = bByte + (u32)(row * 64 + cg * 16);
      asm volatile("global_load_async_to_lds_b128 %0, %1, off"
                   :: "v"(lb), "v"(gb) : "memory");
    }
  };

  v8f acc[2][4];
  #pragma unroll
  for (int i = 0; i < 2; i++)
    #pragma unroll
    for (int j = 0; j < 4; j++)
      #pragma unroll
      for (int e = 0; e < 8; e++) acc[i][j][e] = 0.0f;

  // prologue: stage 0
  issue_stage(0, 0);
  asm volatile("s_wait_asynccnt 0x0" ::: "memory");
  __syncthreads();

  int p = 0;
  for (int k0 = 0; k0 < K; k0 += 32) {
    // prefetch next stage into the other buffer (its readers finished before
    // the barrier that ended the previous iteration)
    if (k0 + 32 < K) issue_stage(p ^ 1, k0 + 32);

    const u32 aDw = (u32)p * 4096u;          // dword index of A stage
    const u32 bDw = aDw + 2048u;             // dword index of B stage
    const int ko  = lhi * 8;                 // A K-base (dwords): 0 / 8
    const int kob = lhi * 8;                 // B K-base (dwords): 0 / 8

    union frag { v16bf v; u32 u[8]; };
    frag fb[4];
    #pragma unroll
    for (int tn = 0; tn < 4; tn++) {
      int nn = wn * 64 + tn * 16 + l16;
      #pragma unroll
      for (int i = 0; i < 8; i++) fb[tn].u[i] = lds_u[bDw + nn * 16 + kob + i];
    }
    #pragma unroll
    for (int tm = 0; tm < 2; tm++) {
      frag fa;
      int mm = wm * 32 + tm * 16 + l16;
      #pragma unroll
      for (int i = 0; i < 4; i++) fa.u[i]     = lds_u[aDw + mm * 16 + ko + i];
      #pragma unroll
      for (int i = 0; i < 4; i++) fa.u[4 + i] = lds_u[aDw + mm * 16 + 4 + ko + i];
      #pragma unroll
      for (int tn = 0; tn < 4; tn++) {
        acc[tm][tn] = __builtin_amdgcn_wmma_f32_16x16x32_bf16(
            false, fa.v, false, fb[tn].v, (short)0, acc[tm][tn], false, false);
      }
    }

    // next-stage async writes complete + all waves' reads of this stage done
    asm volatile("s_wait_asynccnt 0x0" ::: "memory");
    __syncthreads();
    p ^= 1;
  }

  // ---- epilogue: bias / relu, fp32 and/or bf16 stores (C layout §7.12.2) ----
  #pragma unroll
  for (int tm = 0; tm < 2; tm++) {
    #pragma unroll
    for (int tn = 0; tn < 4; tn++) {
      int ng = n0 + wn * 64 + tn * 16 + l16;
      if (ng >= N) continue;
      float bv = bias ? bias[ng] : 0.0f;
      #pragma unroll
      for (int i = 0; i < 8; i++) {
        int mg = m0 + wm * 32 + tm * 16 + lhi * 8 + i;
        float v = acc[tm][tn][i] + bv;
        if (relu) v = v > 0.0f ? v : 0.0f;
        size_t off = (size_t)mg * ldc + ng;
        if (outF) outF[off] = v;
        if (outH) outH[off] = (bf16)v;
      }
    }
  }
}

// ------------------------------------------------------------- masked softmax
__global__ __launch_bounds__(256)
void softmax_row(const float* __restrict__ scores, bf16* __restrict__ attn,
                 int Sk, int masked, float scale)
{
  __shared__ float red[256];
  int q   = blockIdx.x;
  int tid = threadIdx.x;
  int limit = masked ? (q + 1) : Sk;
  const float* row = scores + (size_t)q * Sk;
  bf16* orow = attn + (size_t)q * Sk;

  float mx = -3.4e38f;
  for (int j = tid; j < limit; j += 256) mx = fmaxf(mx, row[j] * scale);
  red[tid] = mx; __syncthreads();
  for (int s = 128; s > 0; s >>= 1) {
    if (tid < s) red[tid] = fmaxf(red[tid], red[tid + s]);
    __syncthreads();
  }
  mx = red[0]; __syncthreads();

  float sum = 0.0f;
  for (int j = tid; j < limit; j += 256) sum += __expf(row[j] * scale - mx);
  red[tid] = sum; __syncthreads();
  for (int s = 128; s > 0; s >>= 1) {
    if (tid < s) red[tid] += red[tid + s];
    __syncthreads();
  }
  float inv = 1.0f / red[0];

  for (int j = tid; j < Sk; j += 256) {
    float p = (j < limit) ? __expf(row[j] * scale - mx) * inv : 0.0f;
    orow[j] = (bf16)p;
  }
}

// --------------------------------------------------------- residual LayerNorm
__global__ __launch_bounds__(256)
void layernorm_res(const float* __restrict__ a, const float* __restrict__ r,
                   const float* __restrict__ g, const float* __restrict__ b,
                   float* __restrict__ outF, bf16* __restrict__ outH, int D)
{
  __shared__ float red[256];
  __shared__ float red2[256];
  int row = blockIdx.x;
  int tid = threadIdx.x;
  const float* ar = a + (size_t)row * D;
  const float* rr = r + (size_t)row * D;

  float s = 0.0f, s2 = 0.0f;
  for (int j = tid; j < D; j += 256) {
    float t = ar[j] + rr[j];
    s += t; s2 += t * t;
  }
  red[tid] = s; red2[tid] = s2; __syncthreads();
  for (int st = 128; st > 0; st >>= 1) {
    if (tid < st) { red[tid] += red[tid + st]; red2[tid] += red2[tid + st]; }
    __syncthreads();
  }
  float mu  = red[0] / (float)D;
  float var = red2[0] / (float)D - mu * mu;
  float inv = rsqrtf(var + 1e-5f);

  for (int j = tid; j < D; j += 256) {
    float t = ar[j] + rr[j];
    float y = (t - mu) * inv * g[j] + b[j];
    outF[(size_t)row * D + j] = y;
    outH[(size_t)row * D + j] = (bf16)y;
  }
}

// ---------------------------------------------------------------------- host
extern "C" void kernel_launch(void* const* d_in, const int* in_sizes, int n_in,
                              void* d_out, int out_size, void* d_ws, size_t ws_size,
                              hipStream_t stream) {
  (void)in_sizes; (void)n_in; (void)out_size; (void)ws_size;

  const float* x    = (const float*)d_in[0];
  const float* enc  = (const float*)d_in[1];
  const float* Wq1  = (const float*)d_in[2];  const float* bq1 = (const float*)d_in[3];
  const float* Wk1  = (const float*)d_in[4];  const float* bk1 = (const float*)d_in[5];
  const float* Wv1  = (const float*)d_in[6];  const float* bv1 = (const float*)d_in[7];
  const float* Wo1  = (const float*)d_in[8];  const float* bo1 = (const float*)d_in[9];
  const float* Wq2  = (const float*)d_in[10]; const float* bq2 = (const float*)d_in[11];
  const float* Wk2  = (const float*)d_in[12]; const float* bk2 = (const float*)d_in[13];
  const float* Wv2  = (const float*)d_in[14]; const float* bv2 = (const float*)d_in[15];
  const float* Wo2  = (const float*)d_in[16]; const float* bo2 = (const float*)d_in[17];
  const float* W1   = (const float*)d_in[18]; const float* b1  = (const float*)d_in[19];
  const float* W2   = (const float*)d_in[20]; const float* b2  = (const float*)d_in[21];
  const float* g1   = (const float*)d_in[22]; const float* be1 = (const float*)d_in[23];
  const float* g2   = (const float*)d_in[24]; const float* be2 = (const float*)d_in[25];
  const float* g3   = (const float*)d_in[26]; const float* be3 = (const float*)d_in[27];
  float* out = (float*)d_out;

  // ---- workspace bump allocator (256B aligned) ----
  char* ws = (char*)d_ws;
  auto alloc = [&](size_t bytes) -> void* {
    void* p = (void*)ws; ws += (bytes + 255) & ~(size_t)255; return p;
  };
  const size_t SD = (size_t)SQ * DMODEL;
  const size_t SS = (size_t)SQ * SQ;

  bf16* xb    = (bf16*)alloc(SD * 2);
  bf16* encb  = (bf16*)alloc(SD * 2);
  bf16* Qb    = (bf16*)alloc(SD * 2);
  bf16* Kb    = (bf16*)alloc(SD * 2);
  bf16* Vb    = (bf16*)alloc(SD * 2);
  bf16* Vt    = (bf16*)alloc(SD * 2);       // V transposed [DMODEL][SQ]
  bf16* Zb    = (bf16*)alloc(SD * 2);
  bf16* Wq1b  = (bf16*)alloc((size_t)DMODEL * DMODEL * 2);
  bf16* Wk1b  = (bf16*)alloc((size_t)DMODEL * DMODEL * 2);
  bf16* Wv1b  = (bf16*)alloc((size_t)DMODEL * DMODEL * 2);
  bf16* Wo1b  = (bf16*)alloc((size_t)DMODEL * DMODEL * 2);
  bf16* Wq2b  = (bf16*)alloc((size_t)DMODEL * DMODEL * 2);
  bf16* Wk2b  = (bf16*)alloc((size_t)DMODEL * DMODEL * 2);
  bf16* Wv2b  = (bf16*)alloc((size_t)DMODEL * DMODEL * 2);
  bf16* Wo2b  = (bf16*)alloc((size_t)DMODEL * DMODEL * 2);
  bf16* W1b   = (bf16*)alloc((size_t)DFF * DMODEL * 2);
  bf16* W2b   = (bf16*)alloc((size_t)DMODEL * DFF * 2);
  float* scoresF = (float*)alloc(SS * 4);
  bf16*  attnH   = (bf16*)alloc(SS * 2);    // reused as FFN hidden (same size)
  float* tmpF    = (float*)alloc(SD * 4);
  float* x1F     = (float*)alloc(SD * 4);
  bf16*  x1H     = (bf16*)alloc(SD * 2);
  float* x2F     = (float*)alloc(SD * 4);
  bf16*  x2H     = (bf16*)alloc(SD * 2);
  bf16*  dummyH  = (bf16*)alloc(SD * 2);

  auto cvt = [&](const float* src, bf16* dst, size_t n) {
    cvt_f32_bf16<<<dim3(1024), dim3(256), 0, stream>>>(src, dst, (int)n);
  };
  auto gemm = [&](const bf16* A, int lda, const bf16* B, int ldb,
                  const float* bias, int relu, float* oF, bf16* oH, int ldc,
                  int M, int N, int K, int causal) {
    dim3 grid((N + 127) / 128, (M + 127) / 128);
    gemm_bf16_wmma<<<grid, dim3(256), 0, stream>>>(A, lda, B, ldb, bias, relu,
                                                   oF, oH, ldc, M, N, K, causal);
  };

  // ---- fp32 -> bf16 staging ----
  cvt(x, xb, SD);      cvt(enc, encb, SD);
  cvt(Wq1, Wq1b, (size_t)DMODEL * DMODEL); cvt(Wk1, Wk1b, (size_t)DMODEL * DMODEL);
  cvt(Wv1, Wv1b, (size_t)DMODEL * DMODEL); cvt(Wo1, Wo1b, (size_t)DMODEL * DMODEL);
  cvt(Wq2, Wq2b, (size_t)DMODEL * DMODEL); cvt(Wk2, Wk2b, (size_t)DMODEL * DMODEL);
  cvt(Wv2, Wv2b, (size_t)DMODEL * DMODEL); cvt(Wo2, Wo2b, (size_t)DMODEL * DMODEL);
  cvt(W1, W1b, (size_t)DFF * DMODEL);      cvt(W2, W2b, (size_t)DMODEL * DFF);

  const float sc = 0.125f;   // 1/sqrt(64)
  const dim3 trGrid(DMODEL / 32, SQ / 32);

  // =================== masked self-attention ===================
  gemm(xb, DMODEL, Wq1b, DMODEL, bq1, 0, nullptr, Qb, DMODEL, SQ, DMODEL, DMODEL, 0);
  gemm(xb, DMODEL, Wk1b, DMODEL, bk1, 0, nullptr, Kb, DMODEL, SQ, DMODEL, DMODEL, 0);
  gemm(xb, DMODEL, Wv1b, DMODEL, bv1, 0, nullptr, Vb, DMODEL, SQ, DMODEL, DMODEL, 0);
  transpose_bf16<<<trGrid, dim3(256), 0, stream>>>(Vb, Vt, SQ, DMODEL);
  for (int h = 0; h < NHEAD; h++) {
    gemm(Qb + h * DKH, DMODEL, Kb + h * DKH, DMODEL, nullptr, 0,
         scoresF, nullptr, SQ, SQ, SQ, DKH, 1);
    softmax_row<<<dim3(SQ), dim3(256), 0, stream>>>(scoresF, attnH, SQ, 1, sc);
    gemm(attnH, SQ, Vt + (size_t)h * DKH * SQ, SQ, nullptr, 0,
         nullptr, Zb + h * DKH, DMODEL, SQ, DKH, SQ, 0);
  }
  gemm(Zb, DMODEL, Wo1b, DMODEL, bo1, 0, tmpF, nullptr, DMODEL, SQ, DMODEL, DMODEL, 0);
  layernorm_res<<<dim3(SQ), dim3(256), 0, stream>>>(tmpF, x, g1, be1, x1F, x1H, DMODEL);

  // =================== encoder-decoder cross attention ===================
  gemm(x1H,  DMODEL, Wq2b, DMODEL, bq2, 0, nullptr, Qb, DMODEL, SQ, DMODEL, DMODEL, 0);
  gemm(encb, DMODEL, Wk2b, DMODEL, bk2, 0, nullptr, Kb, DMODEL, SQ, DMODEL, DMODEL, 0);
  gemm(encb, DMODEL, Wv2b, DMODEL, bv2, 0, nullptr, Vb, DMODEL, SQ, DMODEL, DMODEL, 0);
  transpose_bf16<<<trGrid, dim3(256), 0, stream>>>(Vb, Vt, SQ, DMODEL);
  for (int h = 0; h < NHEAD; h++) {
    gemm(Qb + h * DKH, DMODEL, Kb + h * DKH, DMODEL, nullptr, 0,
         scoresF, nullptr, SQ, SQ, SQ, DKH, 0);
    softmax_row<<<dim3(SQ), dim3(256), 0, stream>>>(scoresF, attnH, SQ, 0, sc);
    gemm(attnH, SQ, Vt + (size_t)h * DKH * SQ, SQ, nullptr, 0,
         nullptr, Zb + h * DKH, DMODEL, SQ, DKH, SQ, 0);
  }
  gemm(Zb, DMODEL, Wo2b, DMODEL, bo2, 0, tmpF, nullptr, DMODEL, SQ, DMODEL, DMODEL, 0);
  layernorm_res<<<dim3(SQ), dim3(256), 0, stream>>>(tmpF, x1F, g2, be2, x2F, x2H, DMODEL);

  // =================== FFN ===================
  bf16* ffn1H = attnH;  // reuse 32MB bf16 buffer
  gemm(x2H, DMODEL, W1b, DMODEL, b1, 1, nullptr, ffn1H, DFF, SQ, DFF, DMODEL, 0);
  gemm(ffn1H, DFF, W2b, DFF, b2, 0, tmpF, nullptr, DMODEL, SQ, DMODEL, DFF, 0);
  layernorm_res<<<dim3(SQ), dim3(256), 0, stream>>>(tmpF, x2F, g3, be3, out, dummyH, DMODEL);
}
